// GNNStack_29858612642389
// MI455X (gfx1250) — compile-verified, matrix operand order checked by reading, not run on hardware
//
#include <hip/hip_runtime.h>
#include <type_traits>

#define NNODES 20000
#define NEDGES 160000
#define DK 512

typedef __bf16 bfx16 __attribute__((ext_vector_type(16)));
typedef __bf16 bfx8  __attribute__((ext_vector_type(8)));
typedef __bf16 bfx4  __attribute__((ext_vector_type(4)));
typedef float  v8f   __attribute__((ext_vector_type(8)));

// ---------------------------------------------------------------------------
// Load a 16-element bf16 WMMA fragment: two contiguous 16B chunks at p[0..7]
// and p[16..23] (CDNA5 16-bit A/B layout; lane supplies row/col + k-half).
// ---------------------------------------------------------------------------
__device__ __forceinline__ bfx16 load_frag(const __bf16* p) {
  const bfx8 lo = *reinterpret_cast<const bfx8*>(p);
  const bfx8 hi = *reinterpret_cast<const bfx8*>(p + 16);
  return __builtin_shufflevector(lo, hi, 0, 1, 2, 3, 4, 5, 6, 7,
                                 8, 9, 10, 11, 12, 13, 14, 15);
}

// ---------------------------------------------------------------------------
// Fused GEMM: C[16 x NCOL] = A1@W1^T (+ A2@W2^T) (+ bias), K = 512, bf16 in,
// fp32 accumulate.
//   MODE 0: row L2-normalize (eps=1e-12) then ReLU -> bf16   (SAGE layer)
//   MODE 1: +bias, store raw -> bf16                         (post_mp linear 1)
//   MODE 2: +bias, fused log_softmax -> fp32                 (post_mp linear 2)
// Workgroup = NCOL/64 waves; wave w covers columns [w*64, w*64+64).
// ---------------------------------------------------------------------------
template <int NCOL, int MODE>
__global__ __launch_bounds__(NCOL / 2) void fused_gemm_kernel(
    const __bf16* __restrict__ A1, const __bf16* __restrict__ WB1,
    const __bf16* __restrict__ A2, const __bf16* __restrict__ WB2,
    const float* __restrict__ bias, void* __restrict__ out_raw) {
  constexpr int NW = NCOL / 64;
  __shared__ float red[16][NW];

  const int lane  = threadIdx.x & 31;
  const int wave  = threadIdx.x >> 5;
  const int rlo   = lane & 15;  // row (A) / col (B,C) within 16x16 tile
  const int khalf = lane >> 4;  // which K half-group this lane holds
  const int kb    = khalf * 8;
  const int rowb  = khalf * 8;  // C rows held by this lane: rowb..rowb+7
  const int m0    = blockIdx.x * 16;

  v8f c[4] = {};

  const __bf16* arow1 = A1 + (size_t)(m0 + rlo) * DK + kb;
  for (int k0 = 0; k0 < DK; k0 += 32) {
    const bfx16 a = load_frag(arow1 + k0);
#pragma unroll
    for (int t = 0; t < 4; ++t) {
      const int col = wave * 64 + t * 16 + rlo;
      const bfx16 b = load_frag(WB1 + (size_t)col * DK + k0 + kb);
      c[t] = __builtin_amdgcn_wmma_f32_16x16x32_bf16(false, a, false, b,
                                                     (short)0, c[t], false, false);
    }
  }
  if (A2 != nullptr) {  // second GEMM accumulated into same C (SAGE layers)
    const __bf16* arow2 = A2 + (size_t)(m0 + rlo) * DK + kb;
    for (int k0 = 0; k0 < DK; k0 += 32) {
      const bfx16 a = load_frag(arow2 + k0);
#pragma unroll
      for (int t = 0; t < 4; ++t) {
        const int col = wave * 64 + t * 16 + rlo;
        const bfx16 b = load_frag(WB2 + (size_t)col * DK + k0 + kb);
        c[t] = __builtin_amdgcn_wmma_f32_16x16x32_bf16(false, a, false, b,
                                                       (short)0, c[t], false, false);
      }
    }
  }

  if (MODE != 0 && bias != nullptr) {
#pragma unroll
    for (int t = 0; t < 4; ++t) {
      const float bv = bias[wave * 64 + t * 16 + rlo];
#pragma unroll
      for (int r = 0; r < 8; ++r) c[t][r] += bv;
    }
  }

  if (MODE == 0) {
    // ---- row-wise L2 norm over all NCOL columns, x/max(||x||,eps), ReLU
    __bf16* out = (__bf16*)out_raw;
    float s[8];
#pragma unroll
    for (int r = 0; r < 8; ++r) {
      float acc = 0.f;
#pragma unroll
      for (int t = 0; t < 4; ++t) acc += c[t][r] * c[t][r];
      for (int off = 1; off < 16; off <<= 1) acc += __shfl_xor(acc, off, 32);
      s[r] = acc;
    }
    if (rlo == 0) {
#pragma unroll
      for (int r = 0; r < 8; ++r) red[rowb + r][wave] = s[r];
    }
    __syncthreads();
#pragma unroll
    for (int r = 0; r < 8; ++r) {
      float tot = 0.f;
      for (int w = 0; w < NW; ++w) tot += red[rowb + r][w];
      const float inv = 1.0f / fmaxf(sqrtf(tot), 1e-12f);
#pragma unroll
      for (int t = 0; t < 4; ++t) {
        const float v = fmaxf(c[t][r] * inv, 0.0f);
        out[(size_t)(m0 + rowb + r) * NCOL + wave * 64 + t * 16 + rlo] = (__bf16)v;
      }
    }
  } else if (MODE == 1) {
    __bf16* out = (__bf16*)out_raw;
#pragma unroll
    for (int r = 0; r < 8; ++r)
#pragma unroll
      for (int t = 0; t < 4; ++t)
        out[(size_t)(m0 + rowb + r) * NCOL + wave * 64 + t * 16 + rlo] =
            (__bf16)c[t][r];
  } else {
    // ---- fused log_softmax across NCOL columns, fp32 output
    float* out = (float*)out_raw;
    float mx[8];
#pragma unroll
    for (int r = 0; r < 8; ++r) {
      float m = c[0][r];
#pragma unroll
      for (int t = 1; t < 4; ++t) m = fmaxf(m, c[t][r]);
      for (int off = 1; off < 16; off <<= 1) m = fmaxf(m, __shfl_xor(m, off, 32));
      mx[r] = m;
    }
    if (rlo == 0) {
#pragma unroll
      for (int r = 0; r < 8; ++r) red[rowb + r][wave] = mx[r];
    }
    __syncthreads();
#pragma unroll
    for (int r = 0; r < 8; ++r) {
      float m = -3.4e38f;
      for (int w = 0; w < NW; ++w) m = fmaxf(m, red[rowb + r][w]);
      mx[r] = m;
    }
    __syncthreads();
    float se[8];
#pragma unroll
    for (int r = 0; r < 8; ++r) {
      float acc = 0.f;
#pragma unroll
      for (int t = 0; t < 4; ++t) acc += expf(c[t][r] - mx[r]);
      for (int off = 1; off < 16; off <<= 1) acc += __shfl_xor(acc, off, 32);
      se[r] = acc;
    }
    if (rlo == 0) {
#pragma unroll
      for (int r = 0; r < 8; ++r) red[rowb + r][wave] = se[r];
    }
    __syncthreads();
#pragma unroll
    for (int r = 0; r < 8; ++r) {
      float tot = 0.f;
      for (int w = 0; w < NW; ++w) tot += red[rowb + r][w];
      const float lse = mx[r] + logf(tot);
#pragma unroll
      for (int t = 0; t < 4; ++t)
        out[(size_t)(m0 + rowb + r) * NCOL + wave * 64 + t * 16 + rlo] =
            c[t][r] - lse;
    }
  }
}

// ---------------------------------------------------------------------------
// One-time fp32 -> bf16 conversion (emb + weight matrices).
// ---------------------------------------------------------------------------
__global__ void cvt_bf16_kernel(const float* __restrict__ in,
                                __bf16* __restrict__ out, int n) {
  const int i = blockIdx.x * blockDim.x + threadIdx.x;
  if (i < n) out[i] = (__bf16)in[i];
}

// ---------------------------------------------------------------------------
// CSR construction (once per call): histogram -> single-block scan -> scatter.
// ---------------------------------------------------------------------------
__global__ void zero_int_kernel(int* p, int n) {
  const int i = blockIdx.x * blockDim.x + threadIdx.x;
  if (i < n) p[i] = 0;
}

__global__ void hist_kernel(const int* __restrict__ dst, int* counts) {
  const int e = blockIdx.x * blockDim.x + threadIdx.x;
  if (e < NEDGES) atomicAdd(&counts[dst[e]], 1);
}

__global__ __launch_bounds__(1024) void scan_kernel(const int* counts, int* rowptr,
                                                    int* cursor) {
  __shared__ int sh[1024];
  const int tid = threadIdx.x;
  int carry = 0;
  for (int base = 0; base < NNODES; base += 1024) {
    const int i = base + tid;
    const int v = (i < NNODES) ? counts[i] : 0;
    sh[tid] = v;
    __syncthreads();
    for (int off = 1; off < 1024; off <<= 1) {
      const int t = (tid >= off) ? sh[tid - off] : 0;
      __syncthreads();
      sh[tid] += t;
      __syncthreads();
    }
    const int incl = sh[tid];
    const int total = sh[1023];
    if (i < NNODES) {
      rowptr[i] = carry + incl - v;
      cursor[i] = carry + incl - v;
    }
    carry += total;
    __syncthreads();
  }
  if (tid == 0) rowptr[NNODES] = carry;
}

__global__ void scatter_kernel(const int* __restrict__ src, const int* __restrict__ dst,
                               int* cursor, int* __restrict__ esrc) {
  const int e = blockIdx.x * blockDim.x + threadIdx.x;
  if (e < NEDGES) {
    const int p = atomicAdd(&cursor[dst[e]], 1);
    esrc[p] = src[e];
  }
}

// ---------------------------------------------------------------------------
// agg[n,:] = sum over incoming edges of h[src,:] (bf16 in, fp32 accumulate,
// bf16 out). One wave per (node, 128-elem chunk); no float atomics.
// ---------------------------------------------------------------------------
__global__ __launch_bounds__(256) void aggregate_kernel(const __bf16* __restrict__ h,
                                                        const int* __restrict__ rowptr,
                                                        const int* __restrict__ esrc,
                                                        __bf16* __restrict__ agg) {
  const int gw   = (blockIdx.x * blockDim.x + threadIdx.x) >> 5;
  const int lane = threadIdx.x & 31;
  const int node = gw >> 2;
  if (node >= NNODES) return;
  const int k = (gw & 3) * 128 + lane * 4;
  float acc[4] = {0.f, 0.f, 0.f, 0.f};
  const int beg = rowptr[node], end = rowptr[node + 1];
  for (int e = beg; e < end; ++e) {
    const int s = esrc[e];
    const bfx4 v = *reinterpret_cast<const bfx4*>(h + (size_t)s * DK + k);
#pragma unroll
    for (int i = 0; i < 4; ++i) acc[i] += (float)v[i];
  }
  bfx4 o;
#pragma unroll
  for (int i = 0; i < 4; ++i) o[i] = (__bf16)acc[i];
  *reinterpret_cast<bfx4*>(agg + (size_t)node * DK + k) = o;
}

// ---------------------------------------------------------------------------
extern "C" void kernel_launch(void* const* d_in, const int* in_sizes, int n_in,
                              void* d_out, int out_size, void* d_ws, size_t ws_size,
                              hipStream_t stream) {
  (void)in_sizes; (void)n_in; (void)out_size; (void)ws_size;
  const int*   ei    = (const int*)d_in[1];
  const float* emb_f = (const float*)d_in[2];
  const float* b1    = (const float*)d_in[10];
  const float* b2    = (const float*)d_in[12];
  const int* src = ei;           // edge_index[0]
  const int* dst = ei + NEDGES;  // edge_index[1]

  char* ws = (char*)d_ws;
  size_t off = 0;
  auto alloc = [&](size_t bytes) -> void* {
    void* p = ws + off;
    off += (bytes + 255) & ~(size_t)255;
    return p;
  };
  const size_t actElems = (size_t)NNODES * DK;
  __bf16* embb = (__bf16*)alloc(actElems * 2);  // bf16 copy of emb
  __bf16* hA   = (__bf16*)alloc(actElems * 2);
  __bf16* hB   = (__bf16*)alloc(actElems * 2);
  __bf16* agg  = (__bf16*)alloc(actElems * 2);  // also reused as post_mp hidden
  // bf16 weights: Wl0,Wr0,Wl1,Wr1,Wl2,Wr2,W1 (512x512), W2 (256x512)
  __bf16* wb[8];
  for (int i = 0; i < 7; ++i) wb[i] = (__bf16*)alloc((size_t)512 * 512 * 2);
  wb[7] = (__bf16*)alloc((size_t)256 * 512 * 2);
  int* rowptr = (int*)alloc((size_t)(NNODES + 1) * sizeof(int));
  int* cursor = (int*)alloc((size_t)NNODES * sizeof(int));
  int* esrc   = (int*)alloc((size_t)NEDGES * sizeof(int));

  // ---- one-time conversions (weights d_in[3..9] order: Wl0,Wr0,Wl1,Wr1,Wl2,Wr2,W1)
  cvt_bf16_kernel<<<(int)((actElems + 255) / 256), 256, 0, stream>>>(emb_f, embb,
                                                                     (int)actElems);
  for (int i = 0; i < 7; ++i)
    cvt_bf16_kernel<<<(512 * 512 + 255) / 256, 256, 0, stream>>>(
        (const float*)d_in[3 + i], wb[i], 512 * 512);
  cvt_bf16_kernel<<<(256 * 512 + 255) / 256, 256, 0, stream>>>(
      (const float*)d_in[11], wb[7], 256 * 512);

  const __bf16 *Wl0 = wb[0], *Wr0 = wb[1], *Wl1 = wb[2], *Wr1 = wb[3];
  const __bf16 *Wl2 = wb[4], *Wr2 = wb[5], *W1 = wb[6], *W2 = wb[7];

  // ---- CSR build (counts aliased into cursor; scan rewrites it with offsets)
  zero_int_kernel<<<(NNODES + 255) / 256, 256, 0, stream>>>(cursor, NNODES);
  hist_kernel<<<(NEDGES + 255) / 256, 256, 0, stream>>>(dst, cursor);
  scan_kernel<<<1, 1024, 0, stream>>>(cursor, rowptr, cursor);
  scatter_kernel<<<(NEDGES + 255) / 256, 256, 0, stream>>>(src, dst, cursor, esrc);

  const int aggBlocks  = (NNODES * 4) / 8;  // 8 waves/block, 4 waves/node
  const int gemmBlocks = NNODES / 16;       // 1250, exact

  // SAGE layer 1: h = relu(normalize(agg@Wr0^T + emb@Wl0^T))
  aggregate_kernel<<<aggBlocks, 256, 0, stream>>>(embb, rowptr, esrc, agg);
  fused_gemm_kernel<512, 0><<<gemmBlocks, 256, 0, stream>>>(agg, Wr0, embb, Wl0,
                                                            nullptr, hA);
  // SAGE layer 2
  aggregate_kernel<<<aggBlocks, 256, 0, stream>>>(hA, rowptr, esrc, agg);
  fused_gemm_kernel<512, 0><<<gemmBlocks, 256, 0, stream>>>(agg, Wr1, hA, Wl1,
                                                            nullptr, hB);
  // SAGE layer 3
  aggregate_kernel<<<aggBlocks, 256, 0, stream>>>(hB, rowptr, esrc, agg);
  fused_gemm_kernel<512, 0><<<gemmBlocks, 256, 0, stream>>>(agg, Wr2, hB, Wl2,
                                                            nullptr, hA);
  // post_mp: Linear(512,512)+b1 -> Linear(512,256)+b2 -> log_softmax
  fused_gemm_kernel<512, 1><<<gemmBlocks, 256, 0, stream>>>(hA, W1, nullptr, nullptr,
                                                            b1, agg);
  fused_gemm_kernel<256, 2><<<gemmBlocks, 128, 0, stream>>>(agg, W2, nullptr, nullptr,
                                                            b2, d_out);
}
